// VNAnchorTransformerBlock_43688407335122
// MI455X (gfx1250) — matrix operand order repaired and checked by reading, not compiled, not run on hardware
//
#include <hip/hip_runtime.h>
#include <hip/hip_bf16.h>
#include <math.h>

// ---------------------------------------------------------------------------
// VN Anchor Transformer Block for MI455X (gfx1250, wave32, WMMA + TDM)
//
// Dims: B=2, C=128, N=256, H=64 (score-MLP hidden), C2=256, L=3*N=768
//
// Matrix work -> v_wmma_f32_16x16x32_f16 (f16 A/B, f32 accum).
// GEMM operand staging -> Tensor Data Mover (tensor_load_to_lds, D# built
// per CDNA5 ISA ch.8), double-buffered in LDS, synced with s_wait_tensorcnt.
// The (B,N,N,C) score tensor never leaves LDS.
// ---------------------------------------------------------------------------

typedef __attribute__((ext_vector_type(16))) _Float16 v16h;
typedef __attribute__((ext_vector_type(8)))  float    v8f;
typedef __attribute__((ext_vector_type(4)))  int      s4i;
typedef __attribute__((ext_vector_type(8)))  int      s8i;

#define DB 2
#define DC 128
#define DN 256
#define DH 64
#define DC2 256
#define DL (3 * DN)

// CDNA5 16-bit A-matrix (16x32) lane/half -> K mapping (ISA 7.12.2)
__device__ __forceinline__ int kmapA(int lane, int h) {
  return (h & 7) + ((h & 8) ? 16 : 0) + ((lane & 16) ? 8 : 0);
}
// CDNA5 16-bit B-matrix (32x16) lane/half -> K mapping
__device__ __forceinline__ int kmapB(int lane, int h) {
  return (h & 7) + ((h & 8) ? 16 : 0) + ((lane & 16) ? 8 : 0);
}

// ---------------------------------------------------------------------------
// TDM 2D tile load: global (f32, row stride stride0 elements) -> LDS.
// D# group0/group1 packed per CDNA5 ISA 8.3/8.4:
//   g0: count=1 | lds_addr | global_addr[56:0] | type=2
//   g1: wg_mask=0, data_size=2 (4B), tensor_dim0/1, tile_dim0/1,
//       tensor_dim0_stride; dim1_stride unused (2D).
// LDS offset = low 32 bits of the flat __shared__ pointer (ISA 10.2 aperture).
// ---------------------------------------------------------------------------
__device__ __forceinline__ void tdm_load_2d_f32(
    unsigned lds_byte_off, const float* tile_gaddr,
    unsigned tile_d0, unsigned tile_d1,
    unsigned tensor_d0, unsigned tensor_d1,
    unsigned long long stride0_elems) {
  unsigned long long ga = (unsigned long long)(uintptr_t)tile_gaddr;
  s4i g0;
  g0[0] = 1;                                            // count=1, user desc
  g0[1] = (int)lds_byte_off;                            // lds_addr
  g0[2] = (int)(unsigned)(ga & 0xFFFFFFFFull);          // global_addr[31:0]
  g0[3] = (int)((unsigned)((ga >> 32) & 0x1FFFFFFull)   // global_addr[56:32]
                | 0x80000000u);                         // type=2 ("image")
  s8i g1;
  g1[0] = (int)(2u << 16);                              // data_size=4B, mask=0
  g1[1] = (int)((tensor_d0 & 0xFFFFu) << 16);           // tensor_dim0[15:0]
  g1[2] = (int)(((tensor_d0 >> 16) & 0xFFFFu) |
                ((tensor_d1 & 0xFFFFu) << 16));         // dim0[31:16]|dim1[15:0]
  g1[3] = (int)(((tensor_d1 >> 16) & 0xFFFFu) |
                ((tile_d0 & 0xFFFFu) << 16));           // dim1[31:16]|tile0
  g1[4] = (int)(tile_d1 & 0xFFFFu);                     // tile1, tile2=0
  g1[5] = (int)(unsigned)(stride0_elems & 0xFFFFFFFFull);
  g1[6] = (int)(unsigned)((stride0_elems >> 32) & 0xFFFFull);
  g1[7] = 0;                                            // dim1_stride hi = 0
  asm volatile("tensor_load_to_lds %0, %1" :: "s"(g0), "s"(g1) : "memory");
}

// ---------------------------------------------------------------------------
// Per-batch GEMM: Y[b] = W @ X[b];  W: Cout x CIN, X: CIN x L.
// One wave per 16x32 output tile; K slabs of 32 staged by TDM (double-
// buffered A and B tiles in LDS), fully unrolled via CIN template.
// ---------------------------------------------------------------------------
template <int CIN>
__global__ void gemm_wmma_kernel(const float* __restrict__ W,
                                 const float* __restrict__ X,
                                 float* __restrict__ Y,
                                 int Cout, int L) {
  const int lane = threadIdx.x & 31;
  const int n0 = blockIdx.x * 32;
  const int m0 = blockIdx.y * 16;
  const int b  = blockIdx.z;
  const float* __restrict__ Xb = X + (size_t)b * CIN * L;
  float* __restrict__ Yb = Y + (size_t)b * Cout * L;

  __shared__ float sA[2][16 * 32];  // A slab: 16 rows x 32 k (row-major)
  __shared__ float sB[2][32 * 32];  // B slab: 32 k x 32 n (row-major)

  constexpr int nSlab = CIN / 32;

  // Prologue: stage slab 0 (A then B).
  tdm_load_2d_f32((unsigned)(uintptr_t)&sA[0][0], &W[(size_t)m0 * CIN],
                  32, 16, (unsigned)CIN, (unsigned)(Cout - m0),
                  (unsigned long long)CIN);
  tdm_load_2d_f32((unsigned)(uintptr_t)&sB[0][0], &Xb[n0],
                  32, 32, (unsigned)(L - n0), (unsigned)CIN,
                  (unsigned long long)L);

  v8f acc0 = {}, acc1 = {};
#pragma unroll
  for (int s = 0; s < nSlab; ++s) {
    const int buf = s & 1;
    if (s + 1 < nSlab) {
      const int k1 = (s + 1) * 32;
      tdm_load_2d_f32((unsigned)(uintptr_t)&sA[buf ^ 1][0],
                      &W[(size_t)m0 * CIN + k1],
                      32, 16, (unsigned)(CIN - k1), (unsigned)(Cout - m0),
                      (unsigned long long)CIN);
      tdm_load_2d_f32((unsigned)(uintptr_t)&sB[buf ^ 1][0],
                      &Xb[(size_t)k1 * L + n0],
                      32, 32, (unsigned)(L - n0), (unsigned)(CIN - k1),
                      (unsigned long long)L);
      // 2 outstanding for slab s+1; slab s complete when TENSORcnt <= 2.
      asm volatile("s_wait_tensorcnt 0x2" ::: "memory");
    } else {
      asm volatile("s_wait_tensorcnt 0x0" ::: "memory");
    }

    v16h a;
    const int rA = lane & 15;
#pragma unroll
    for (int h = 0; h < 16; ++h)
      a[h] = (_Float16)sA[buf][rA * 32 + kmapA(lane, h)];

    {
      v16h bb;
      const int cB = lane & 15;
#pragma unroll
      for (int h = 0; h < 16; ++h)
        bb[h] = (_Float16)sB[buf][kmapB(lane, h) * 32 + cB];
      acc0 = __builtin_amdgcn_wmma_f32_16x16x32_f16(
          false, a, false, bb, (short)0, acc0, false, false);
    }
    {
      v16h bb;
      const int cB = 16 + (lane & 15);
#pragma unroll
      for (int h = 0; h < 16; ++h)
        bb[h] = (_Float16)sB[buf][kmapB(lane, h) * 32 + cB];
      acc1 = __builtin_amdgcn_wmma_f32_16x16x32_f16(
          false, a, false, bb, (short)0, acc1, false, false);
    }
  }

  const int rbase = m0 + ((lane & 16) ? 8 : 0);
  const int c0 = n0 + (lane & 15);
#pragma unroll
  for (int r = 0; r < 8; ++r) {
    Yb[(size_t)(rbase + r) * L + c0] = acc0[r];
    Yb[(size_t)(rbase + r) * L + c0 + 16] = acc1[r];
  }
}

// ---------------------------------------------------------------------------
// Fused channel-wise-softmax attention, one block per (query i, batch b).
// ---------------------------------------------------------------------------
__global__ void attn_kernel(const float* __restrict__ q,
                            const float* __restrict__ k,
                            const float* __restrict__ v,
                            const float* __restrict__ pos,
                            const float* __restrict__ W1,
                            const float* __restrict__ b1,
                            const float* __restrict__ W2,
                            const float* __restrict__ b2,
                            float* __restrict__ ao) {
  const int i = blockIdx.x;
  const int b = blockIdx.y;
  const int t = threadIdx.x;  // 0..255 (8 waves)

  extern __shared__ unsigned char smem_raw[];
  float* sQ   = (float*)smem_raw;          // C*3   = 384 floats
  float* sPI  = sQ + DC * 3;               // 4
  float* sW1  = sPI + 4;                   // H*2   = 128
  float* sB1  = sW1 + DH * 2;              // H     = 64
  float* sB2  = sB1 + DH;                  // C     = 128
  _Float16* sW2h = (_Float16*)(sB2 + DC);  // C*H   = 8192 halves (W2[c][k])
  _Float16* sH   = sW2h + DC * DH;         // N*H   = 16384 halves
  float* sS  = (float*)(sH + DN * DH);     // N*C   = 32768 floats

  // ---- Phase 1: stage shared operands ----
  for (int idx = t; idx < DC * 3; idx += 256) {
    int c = idx / 3, d = idx % 3;
    sQ[idx] = q[((size_t)(b * DC + c) * 3 + d) * DN + i];
  }
  if (t < 3) sPI[t] = pos[((size_t)b * DN + i) * 3 + t];
  for (int idx = t; idx < DH * 2; idx += 256) sW1[idx] = W1[idx];
  if (t < DH) sB1[t] = b1[t];
  if (t < DC) sB2[t] = b2[t];
  for (int idx = t; idx < DC * DH; idx += 256) sW2h[idx] = (_Float16)W2[idx];
  __syncthreads();

  // ---- Phase 2: per-key relative stats + score-MLP hidden ----
  for (int j = t; j < DN; j += 256) {
    float dx = sPI[0] - pos[((size_t)b * DN + j) * 3 + 0];
    float dy = sPI[1] - pos[((size_t)b * DN + j) * 3 + 1];
    float dz = sPI[2] - pos[((size_t)b * DN + j) * 3 + 2];
    float psq = dx * dx + dy * dy + dz * dz;
    float rp = (psq > 0.f) ? sqrtf(psq) : 0.f;   // safe_norm

    float rm = 0.f;
    for (int c = 0; c < DC; ++c) {
      float e0 = sQ[c * 3 + 0] - k[((size_t)(b * DC + c) * 3 + 0) * DN + j];
      float e1 = sQ[c * 3 + 1] - k[((size_t)(b * DC + c) * 3 + 1) * DN + j];
      float e2 = sQ[c * 3 + 2] - k[((size_t)(b * DC + c) * 3 + 2) * DN + j];
      float s = e0 * e0 + e1 * e1 + e2 * e2;
      rm += (s > 0.f) ? sqrtf(s) : 0.f;
    }
    rm *= (1.0f / (float)DC);

    for (int kk = 0; kk < DH; ++kk) {
      float hv = sW1[kk * 2 + 0] * rm + sW1[kk * 2 + 1] * rp + sB1[kk];
      hv = (hv > 0.f) ? hv : 0.2f * hv;  // LeakyReLU(0.2)
      sH[j * DH + kk] = (_Float16)hv;
    }
  }
  __syncthreads();

  // ---- Phase 3: S = H @ W2^T + b2 via WMMA (16 j-tiles x 8 c-tiles) ----
  {
    const int wave = t >> 5;
    const int lane = t & 31;
    for (int tile = wave; tile < 16 * 8; tile += 8) {  // same trips all waves
      const int jt = tile >> 3;
      const int ct = tile & 7;
      const int rowA = jt * 16 + (lane & 15);
      const int colB = ct * 16 + (lane & 15);
      v8f acc = {};
#pragma unroll
      for (int k0 = 0; k0 < DH; k0 += 32) {
        v16h a, bb;
#pragma unroll
        for (int h = 0; h < 16; ++h) {
          a[h]  = sH[rowA * DH + k0 + kmapA(lane, h)];
          // B[k][n] = W2^T[k][n] = W2[n][k]
          bb[h] = sW2h[colB * DH + k0 + kmapB(lane, h)];
        }
        acc = __builtin_amdgcn_wmma_f32_16x16x32_f16(
            false, a, false, bb, (short)0, acc, false, false);
      }
      const int rbase = jt * 16 + ((lane & 16) ? 8 : 0);
      const float bias = sB2[colB];
#pragma unroll
      for (int r = 0; r < 8; ++r) {
        sS[(rbase + r) * DC + colB] = acc[r] + bias;
      }
    }
  }
  __syncthreads();

  // ---- Phase 4: softmax over j per channel, weighted sum of v ----
  if (t < DC) {
    const int c = t;
    float m = -3.4e38f;
    for (int j = 0; j < DN; ++j) m = fmaxf(m, sS[j * DC + c]);
    float Z = 0.f, o0 = 0.f, o1 = 0.f, o2 = 0.f;
    for (int j = 0; j < DN; ++j) {
      float e = __expf(sS[j * DC + c] - m);
      Z += e;
      o0 += e * v[((size_t)(b * DC + c) * 3 + 0) * DN + j];
      o1 += e * v[((size_t)(b * DC + c) * 3 + 1) * DN + j];
      o2 += e * v[((size_t)(b * DC + c) * 3 + 2) * DN + j];
    }
    const float inv = 1.0f / Z;
    ao[((size_t)(b * DC + c) * 3 + 0) * DN + i] = o0 * inv;
    ao[((size_t)(b * DC + c) * 3 + 1) * DN + i] = o1 * inv;
    ao[((size_t)(b * DC + c) * 3 + 2) * DN + i] = o2 * inv;
  }
}

// ---------------------------------------------------------------------------
// Residual add Y = X + A fused with covariance-statistic reduction.
// ---------------------------------------------------------------------------
__global__ void resid_stats_kernel(const float* __restrict__ X,
                                   const float* __restrict__ A,
                                   float* __restrict__ Y,
                                   float* __restrict__ statsA) {
  const int b = blockIdx.x;
  const int t = threadIdx.x;
  __shared__ float red[9][256];
  float s[9];
#pragma unroll
  for (int r = 0; r < 9; ++r) s[r] = 0.f;

  const size_t base = (size_t)b * DC * 3 * DN;
  for (int idx = t; idx < DC * DN; idx += 256) {
    const int c = idx / DN, n = idx % DN;
    const size_t o = base + (size_t)c * 3 * DN + n;
    float y0 = X[o] + A[o];
    float y1 = X[o + DN] + A[o + DN];
    float y2 = X[o + 2 * DN] + A[o + 2 * DN];
    Y[o] = y0; Y[o + DN] = y1; Y[o + 2 * DN] = y2;
    s[0] += y0; s[1] += y1; s[2] += y2;
    s[3] += y0 * y0; s[4] += y0 * y1; s[5] += y0 * y2;
    s[6] += y1 * y1; s[7] += y1 * y2; s[8] += y2 * y2;
  }
#pragma unroll
  for (int r = 0; r < 9; ++r) red[r][t] = s[r];
  __syncthreads();
  for (int off = 128; off > 0; off >>= 1) {
    if (t < off) {
#pragma unroll
      for (int r = 0; r < 9; ++r) red[r][t] += red[r][t + off];
    }
    __syncthreads();
  }
  if (t == 0) {
#pragma unroll
    for (int r = 0; r < 9; ++r) statsA[b * 12 + r] = red[r][0];
  }
}

// ---------------------------------------------------------------------------
// ZCA solve: cov from raw sums, inverse sqrt via Newton-Schulz.
// ---------------------------------------------------------------------------
__global__ void zca_solve_kernel(const float* __restrict__ statsA,
                                 float* __restrict__ statsB) {
  const int b = blockIdx.x;
  if (threadIdx.x != 0) return;
  const float M = (float)(DC * DN);
  const float invM = 1.0f / M;
  const float mu0 = statsA[b * 12 + 0] * invM;
  const float mu1 = statsA[b * 12 + 1] * invM;
  const float mu2 = statsA[b * 12 + 2] * invM;
  const float dn = 1.0f / (M + 1e-6f);

  float A[3][3];
  A[0][0] = (statsA[b * 12 + 3] - M * mu0 * mu0) * dn + 1e-5f;
  A[0][1] = (statsA[b * 12 + 4] - M * mu0 * mu1) * dn;
  A[0][2] = (statsA[b * 12 + 5] - M * mu0 * mu2) * dn;
  A[1][1] = (statsA[b * 12 + 6] - M * mu1 * mu1) * dn + 1e-5f;
  A[1][2] = (statsA[b * 12 + 7] - M * mu1 * mu2) * dn;
  A[2][2] = (statsA[b * 12 + 8] - M * mu2 * mu2) * dn + 1e-5f;
  A[1][0] = A[0][1]; A[2][0] = A[0][2]; A[2][1] = A[1][2];

  const float c = A[0][0] + A[1][1] + A[2][2];  // spectral bound for SPD
  const float invc = 1.0f / c;
  float Yn[3][3], Zn[3][3], T[3][3], Tm[3][3];
#pragma unroll
  for (int r = 0; r < 3; ++r)
#pragma unroll
    for (int q2 = 0; q2 < 3; ++q2) {
      Yn[r][q2] = A[r][q2] * invc;
      Zn[r][q2] = (r == q2) ? 1.f : 0.f;
    }
  for (int it = 0; it < 20; ++it) {
#pragma unroll
    for (int r = 0; r < 3; ++r)
#pragma unroll
      for (int q2 = 0; q2 < 3; ++q2) {
        float acc = 0.f;
#pragma unroll
        for (int p = 0; p < 3; ++p) acc += Zn[r][p] * Yn[p][q2];
        T[r][q2] = ((r == q2) ? 1.5f : 0.f) - 0.5f * acc;
      }
#pragma unroll
    for (int r = 0; r < 3; ++r)
#pragma unroll
      for (int q2 = 0; q2 < 3; ++q2) {
        float acc = 0.f;
#pragma unroll
        for (int p = 0; p < 3; ++p) acc += Yn[r][p] * T[p][q2];
        Tm[r][q2] = acc;
      }
#pragma unroll
    for (int r = 0; r < 3; ++r)
#pragma unroll
      for (int q2 = 0; q2 < 3; ++q2) Yn[r][q2] = Tm[r][q2];
#pragma unroll
    for (int r = 0; r < 3; ++r)
#pragma unroll
      for (int q2 = 0; q2 < 3; ++q2) {
        float acc = 0.f;
#pragma unroll
        for (int p = 0; p < 3; ++p) acc += T[r][p] * Zn[p][q2];
        Tm[r][q2] = acc;
      }
#pragma unroll
    for (int r = 0; r < 3; ++r)
#pragma unroll
      for (int q2 = 0; q2 < 3; ++q2) Zn[r][q2] = Tm[r][q2];
  }
  const float sc = rsqrtf(c);  // A^{-1/2} = Zn / sqrt(c)
  statsB[b * 12 + 0] = mu0;
  statsB[b * 12 + 1] = mu1;
  statsB[b * 12 + 2] = mu2;
#pragma unroll
  for (int r = 0; r < 3; ++r)
#pragma unroll
    for (int q2 = 0; q2 < 3; ++q2)
      statsB[b * 12 + 3 + r * 3 + q2] = Zn[r][q2] * sc;
}

// ---------------------------------------------------------------------------
// ZCA apply: out = gamma[c] * Wm @ (y - mu)
// ---------------------------------------------------------------------------
__global__ void zca_apply_kernel(const float* __restrict__ Y,
                                 const float* __restrict__ statsB,
                                 const float* __restrict__ gamma,
                                 float* __restrict__ Out) {
  const int idx = blockIdx.x * blockDim.x + threadIdx.x;  // B*C*N
  const int b = idx / (DC * DN);
  const int r = idx % (DC * DN);
  const int c = r / DN;
  const int n = r % DN;
  const float* st = statsB + b * 12;
  const size_t o = ((size_t)(b * DC + c) * 3) * DN + n;
  const float y0 = Y[o] - st[0];
  const float y1 = Y[o + DN] - st[1];
  const float y2 = Y[o + 2 * DN] - st[2];
  const float g = gamma[c];
  Out[o]          = g * (st[3] * y0 + st[4] * y1 + st[5] * y2);
  Out[o + DN]     = g * (st[6] * y0 + st[7] * y1 + st[8] * y2);
  Out[o + 2 * DN] = g * (st[9] * y0 + st[10] * y1 + st[11] * y2);
}

// ---------------------------------------------------------------------------
// VN LeakyReLU (ffn), in-place over P.
// ---------------------------------------------------------------------------
__global__ void vnlrelu_kernel(float* __restrict__ P,
                               const float* __restrict__ D) {
  const int idx = blockIdx.x * blockDim.x + threadIdx.x;  // B*C2*N
  const int b = idx / (DC2 * DN);
  const int r = idx % (DC2 * DN);
  const int c = r / DN;
  const int n = r % DN;
  const size_t o = ((size_t)(b * DC2 + c) * 3) * DN + n;
  const float p0 = P[o], p1 = P[o + DN], p2 = P[o + 2 * DN];
  const float d0 = D[o], d1 = D[o + DN], d2 = D[o + 2 * DN];
  const float dot = p0 * d0 + p1 * d1 + p2 * d2;
  const float dsq = d0 * d0 + d1 * d1 + d2 * d2;
  const float s = dot / (dsq + 1e-6f);
  const float ns = 0.1f;
  float h0, h1, h2;
  if (dot >= 0.f) {
    h0 = p0; h1 = p1; h2 = p2;
  } else {
    h0 = ns * p0 + (1.f - ns) * (p0 - s * d0);
    h1 = ns * p1 + (1.f - ns) * (p1 - s * d1);
    h2 = ns * p2 + (1.f - ns) * (p2 - s * d2);
  }
  P[o] = h0; P[o + DN] = h1; P[o + 2 * DN] = h2;
}

// ---------------------------------------------------------------------------
extern "C" void kernel_launch(void* const* d_in, const int* in_sizes, int n_in,
                              void* d_out, int out_size, void* d_ws, size_t ws_size,
                              hipStream_t stream) {
  const float* x      = (const float*)d_in[0];
  const float* pos    = (const float*)d_in[1];
  const float* Wq     = (const float*)d_in[2];
  const float* Wk     = (const float*)d_in[3];
  const float* Wv     = (const float*)d_in[4];
  const float* Wo     = (const float*)d_in[5];
  const float* W1     = (const float*)d_in[6];
  const float* b1     = (const float*)d_in[7];
  const float* W2     = (const float*)d_in[8];
  const float* b2     = (const float*)d_in[9];
  const float* gamma1 = (const float*)d_in[10];
  const float* Wfeat  = (const float*)d_in[11];
  const float* Wdir   = (const float*)d_in[12];
  const float* Wffn2  = (const float*)d_in[13];
  const float* gamma2 = (const float*)d_in[14];

  const size_t szX = (size_t)DB * DC * 3 * DN;   // 196608 floats
  const size_t szP = (size_t)DB * DC2 * 3 * DN;  // 393216 floats
  float* ws = (float*)d_ws;
  float* bq  = ws;            // q        -> later Wo out (a), later ffn out (f)
  float* bk  = bq + szX;      // k        -> later residual y / y2
  float* bv  = bk + szX;      // v        -> later x1 (post ZCA-1)
  float* bao = bv + szX;      // attention output (pre-Wo)
  float* bp  = bao + szX;     // ffn p (in-place h)
  float* bd  = bp + szP;      // ffn d
  float* statsA = bd + szP;   // B*12 raw sums
  float* statsB = statsA + DB * 12;  // B*12: mu + Wm
  float* ba  = bq;   // Wo output reuses q
  float* by  = bk;   // residual y reuses k
  float* bx1 = bv;   // ZCA-1 output reuses v
  float* bf  = bq;   // ffn output reuses a

  const size_t smemAttn =
      (size_t)(DC * 3 + 4 + DH * 2 + DH + DC) * sizeof(float) +
      (size_t)(DC * DH + DN * DH) * sizeof(_Float16) +
      (size_t)(DN * DC) * sizeof(float);  // ~183 KB, < 320 KB/WGP

  dim3 gC(DL / 32, DC / 16, DB);    // 24 x 8 x 2
  dim3 gC2(DL / 32, DC2 / 16, DB);  // 24 x 16 x 2

  // --- attention: q,k,v projections ---
  gemm_wmma_kernel<DC><<<gC, 32, 0, stream>>>(Wq, x, bq, DC, DL);
  gemm_wmma_kernel<DC><<<gC, 32, 0, stream>>>(Wk, x, bk, DC, DL);
  gemm_wmma_kernel<DC><<<gC, 32, 0, stream>>>(Wv, x, bv, DC, DL);

  // --- fused channel-wise softmax attention ---
  attn_kernel<<<dim3(DN, DB), 256, smemAttn, stream>>>(
      bq, bk, bv, pos, W1, b1, W2, b2, bao);

  // --- output projection ---
  gemm_wmma_kernel<DC><<<gC, 32, 0, stream>>>(Wo, bao, ba, DC, DL);

  // --- residual + ZCA layernorm 1 ---
  resid_stats_kernel<<<DB, 256, 0, stream>>>(x, ba, by, statsA);
  zca_solve_kernel<<<DB, 32, 0, stream>>>(statsA, statsB);
  zca_apply_kernel<<<(DB * DC * DN) / 256, 256, 0, stream>>>(
      by, statsB, gamma1, bx1);

  // --- FFN ---
  gemm_wmma_kernel<DC><<<gC2, 32, 0, stream>>>(Wfeat, bx1, bp, DC2, DL);
  gemm_wmma_kernel<DC><<<gC2, 32, 0, stream>>>(Wdir, bx1, bd, DC2, DL);
  vnlrelu_kernel<<<(DB * DC2 * DN) / 256, 256, 0, stream>>>(bp, bd);
  gemm_wmma_kernel<DC2><<<gC, 32, 0, stream>>>(Wffn2, bp, bf, DC, DL);

  // --- residual + ZCA layernorm 2 -> d_out ---
  resid_stats_kernel<<<DB, 256, 0, stream>>>(bx1, bf, bk, statsA);
  zca_solve_kernel<<<DB, 32, 0, stream>>>(statsA, statsB);
  zca_apply_kernel<<<(DB * DC * DN) / 256, 256, 0, stream>>>(
      bk, statsB, gamma2, (float*)d_out);
}
